// InterPhaseAttention_21380347199837
// MI455X (gfx1250) — matrix operand order repaired and checked by reading, not compiled, not run on hardware
//
#include <hip/hip_runtime.h>
#include <hip/hip_bf16.h>

// ---------------------------------------------------------------------------
// InterPhaseAttention for MI455X (gfx1250, wave32, WMMA + TDM)
// B=4, T=2048, C=512, H=8, dk=64, phases=3
// ---------------------------------------------------------------------------

typedef __attribute__((ext_vector_type(16))) _Float16 v16h;
typedef __attribute__((ext_vector_type(8)))  _Float16 v8h;
typedef __attribute__((ext_vector_type(8)))  float    v8f;
typedef __attribute__((ext_vector_type(4)))  unsigned int v4u;
typedef __attribute__((ext_vector_type(8)))  int      v8i;
typedef __attribute__((ext_vector_type(4)))  int      v4i;

#define BATCH   4
#define SEQ     2048
#define CDIM    512
#define NHEADS  8
#define DK      64
#define MROWS   (BATCH * SEQ)        // 8192
#define LN_EPS  1e-5f
#define SCALE   0.125f               // 1/sqrt(64)
#define KTILES  16                   // CDIM / 32

#if defined(__HIP_DEVICE_COMPILE__) && \
    __has_builtin(__builtin_amdgcn_tensor_load_to_lds) && \
    __has_builtin(__builtin_amdgcn_s_wait_tensorcnt)
#define HAVE_TDM 1
#endif

// ---------------------------------------------------------------------------
// Fragment layouts used by this code (contiguous K, wave32):
//   A-tile (16x32 f16): lane l (0..15): M=l, halves j -> K=j ; lane l+16: M=l, K=16+j
//   B-tile (32x16 f16): lane l (0..15): N=l, K=j ; lane l+16: N=l, K=16+j
// Each tile = 512 halves = 1024B, lane stride 16 halves (32B) -> one v16h load.
// ---------------------------------------------------------------------------

__device__ __forceinline__ int a_k_of(int j, int hi) {
    return (hi << 4) + j;
}

#if defined(HAVE_TDM)
// 1-D TDM copy: n8 x 8-byte elements from global -> LDS (TENSORcnt tracked).
// D# per ISA ch.8: group0 {count=1, lds_addr, global_addr, type=2},
// group1 {data_size=8B, tensor_dim0 (large), tile_dim0=n8}; groups 2/3 zero.
// This toolchain exposes the 6-arg builtin:
//   (uint32x4 g0, int32x8 g1, int32x4 g2, int32x4 g3, int32x8 extra, i32 cpol)
__device__ __forceinline__ void tdm_load_1d(const void* gsrc, void* lds_dst, int n8)
{
    unsigned long long ga = (unsigned long long)(uintptr_t)gsrc;
    unsigned int la = (unsigned int)(uintptr_t)lds_dst;   // LDS byte offset
    v4u g0;
    g0[0] = 1u;                                           // count=1 descriptor
    g0[1] = la;                                           // lds_addr
    g0[2] = (unsigned int)ga;                             // global_addr[31:0]
    g0[3] = (unsigned int)(((ga >> 32) & 0x1FFFFFFu) | (2u << 30)); // [56:32]|type=2
    const unsigned int td0 = 1u << 20;                    // big tensor_dim0 (no OOB clip)
    v8i g1;
    g1[0] = (int)(3u << 16);                              // data_size=3 (8 bytes)
    g1[1] = (int)((td0 & 0xFFFFu) << 16);                 // tensor_dim0[15:0] @63:48
    g1[2] = (int)((td0 >> 16) & 0xFFFFu);                 // tensor_dim0[31:16] @79:64
    g1[3] = (int)(((unsigned int)n8 & 0xFFFFu) << 16);    // tile_dim0 @127:112
    g1[4] = 0;                                            // tile_dim1/2 unused
    g1[5] = (int)td0;                                     // tensor_dim0_stride[31:0]
    g1[6] = 0;
    g1[7] = 0;
    v4i z4 = {0, 0, 0, 0};
    v8i z8 = {0, 0, 0, 0, 0, 0, 0, 0};
    __builtin_amdgcn_tensor_load_to_lds(g0, g1, z4, z4, z8, 0);
}
#endif

// ---- pack a full row-major f32 matrix [Mt*16 x Kt*32] into A tiles ---------
__global__ __launch_bounds__(128) void pack_a_full(
    const float* __restrict__ src, _Float16* __restrict__ dst,
    int Ktiles, int ld)
{
    int lane = threadIdx.x & 31;
    int warp = threadIdx.x >> 5;
    int tile = blockIdx.x * 4 + warp;
    int mt = tile / Ktiles, kt = tile % Ktiles;
    int lo = lane & 15, hi = lane >> 4;
    int m = mt * 16 + lo;
    v16h frag;
#pragma unroll
    for (int j = 0; j < 16; ++j) {
        int k = kt * 32 + a_k_of(j, hi);
        frag[j] = (_Float16)src[(size_t)m * ld + k];
    }
    ((v16h*)dst)[(size_t)tile * 32 + lane] = frag;
}

// ---- pack weight W[n,k] (torch Linear) as B tiles of W^T: B[k,n]=W[n,k] ----
__global__ __launch_bounds__(128) void pack_b_weight(
    const float* __restrict__ W, _Float16* __restrict__ dst)
{
    int lane = threadIdx.x & 31;
    int warp = threadIdx.x >> 5;
    int tile = blockIdx.x * 4 + warp;       // kt*32 + nt, Ntiles=32
    int kt = tile >> 5, nt = tile & 31;
    int lo = lane & 15, hi = lane >> 4;
    int n = nt * 16 + lo;
    v16h frag;
#pragma unroll
    for (int j = 0; j < 16; ++j) {
        int k = kt * 32 + hi * 16 + j;
        frag[j] = (_Float16)W[(size_t)n * CDIM + k];
    }
    ((v16h*)dst)[(size_t)tile * 32 + lane] = frag;
}

// ---- per-(b,h) pack Q [T x dk] as A tiles: tile = mt*2 + kt ----------------
__global__ __launch_bounds__(128) void pack_q_heads(
    const float* __restrict__ qf, _Float16* __restrict__ dst)
{
    int lane = threadIdx.x & 31;
    int warp = threadIdx.x >> 5;
    int bh = blockIdx.y, b = bh >> 3, h = bh & 7;
    int tile = blockIdx.x * 4 + warp;       // 0..255
    int mt = tile >> 1, kt = tile & 1;
    int lo = lane & 15, hi = lane >> 4;
    int t = mt * 16 + lo;
    v16h frag;
#pragma unroll
    for (int j = 0; j < 16; ++j) {
        int d = kt * 32 + a_k_of(j, hi);
        frag[j] = (_Float16)qf[((size_t)(b * SEQ + t)) * CDIM + h * DK + d];
    }
    ((v16h*)dst)[((size_t)bh * 256 + tile) * 32 + lane] = frag;
}

// ---- per-(b,h) pack K^T [dk x T] as B tiles: tile = ktile*128 + ntile ------
__global__ __launch_bounds__(128) void pack_kT_heads(
    const float* __restrict__ kf, _Float16* __restrict__ dst)
{
    int lane = threadIdx.x & 31;
    int warp = threadIdx.x >> 5;
    int bh = blockIdx.y, b = bh >> 3, h = bh & 7;
    int tile = blockIdx.x * 4 + warp;       // 0..255
    int ktile = tile >> 7, nt = tile & 127;
    int lo = lane & 15, hi = lane >> 4;
    int t = nt * 16 + lo;                   // N = key position
    v16h frag;
#pragma unroll
    for (int j = 0; j < 16; ++j) {
        int d = ktile * 32 + hi * 16 + j;   // K = feature dim
        frag[j] = (_Float16)kf[((size_t)(b * SEQ + t)) * CDIM + h * DK + d];
    }
    ((v16h*)dst)[((size_t)bh * 256 + tile) * 32 + lane] = frag;
}

// ---- per-(b,h) pack V [T x dk] as B tiles: tile = t32*4 + ntile ------------
__global__ __launch_bounds__(128) void pack_v_heads(
    const float* __restrict__ vf, _Float16* __restrict__ dst)
{
    int lane = threadIdx.x & 31;
    int warp = threadIdx.x >> 5;
    int bh = blockIdx.y, b = bh >> 3, h = bh & 7;
    int tile = blockIdx.x * 4 + warp;       // 0..255
    int t32 = tile >> 2, nt = tile & 3;
    int lo = lane & 15, hi = lane >> 4;
    int d = nt * 16 + lo;                   // N = feature dim
    v16h frag;
#pragma unroll
    for (int j = 0; j < 16; ++j) {
        int t = t32 * 32 + hi * 16 + j;     // K = key position
        frag[j] = (_Float16)vf[((size_t)(b * SEQ + t)) * CDIM + h * DK + d];
    }
    ((v16h*)dst)[((size_t)bh * 256 + tile) * 32 + lane] = frag;
}

// ---------------------------------------------------------------------------
// Tiled WMMA GEMM: out[M,512] = Apack x Bpack (+bias) (+residual), K = 512.
// Block = 4 waves; wave computes 16x64. B panel (4KB) staged in LDS by the
// Tensor Data Mover, double-buffered: TDM loads panel k+1 while waves compute
// panel k. A fragments are register-prefetched one K-step ahead.
// ---------------------------------------------------------------------------
__global__ __launch_bounds__(128) void gemm_wmma(
    const _Float16* __restrict__ Ap, const _Float16* __restrict__ Bp,
    const float* __restrict__ bias, const float* __restrict__ res,
    float* __restrict__ out)
{
    const int Ntiles = CDIM / 16;           // 32
    __shared__ __align__(16) _Float16 bpanel[2][4 * 512];   // 2 x 4KB
    int lane = threadIdx.x & 31;
    int warp = threadIdx.x >> 5;
    int mt  = blockIdx.y * 4 + warp;        // 16-row tile
    int nt0 = blockIdx.x * 4;               // 16-col tile base
    v8f acc[4] = {};
    const v16h* A = (const v16h*)Ap;
    const v16h* B = (const v16h*)Bp;

    v16h a_cur = A[((size_t)mt * KTILES + 0) * 32 + lane];

#if defined(HAVE_TDM)
    // prologue: panels 0 and 1 in flight
    if (warp == 0) {
        tdm_load_1d(Bp + ((size_t)0 * Ntiles + nt0) * 512, &bpanel[0][0], 512);
        tdm_load_1d(Bp + ((size_t)1 * Ntiles + nt0) * 512, &bpanel[1][0], 512);
    }
    for (int kt = 0; kt < KTILES; ++kt) {
        v16h a_nxt = a_cur;
        if (kt + 1 < KTILES)
            a_nxt = A[((size_t)mt * KTILES + kt + 1) * 32 + lane];
        if (warp == 0) {
            if (kt < KTILES - 1) __builtin_amdgcn_s_wait_tensorcnt(1);
            else                 __builtin_amdgcn_s_wait_tensorcnt(0);
        }
        __syncthreads();                    // panel kt visible to all waves
        const v16h* Bl = (const v16h*)&bpanel[kt & 1][0];
#pragma unroll
        for (int i = 0; i < 4; ++i) {
            v16h b = Bl[i * 32 + lane];
            acc[i] = __builtin_amdgcn_wmma_f32_16x16x32_f16(
                false, a_cur, false, b, (short)0, acc[i], false, false);
        }
        __syncthreads();                    // everyone done reading buffer kt&1
        if (warp == 0 && kt + 2 < KTILES)
            tdm_load_1d(Bp + ((size_t)(kt + 2) * Ntiles + nt0) * 512,
                        &bpanel[kt & 1][0], 512);
        a_cur = a_nxt;
    }
#else
    for (int kt = 0; kt < KTILES; ++kt) {
        v16h a_nxt = a_cur;
        if (kt + 1 < KTILES)
            a_nxt = A[((size_t)mt * KTILES + kt + 1) * 32 + lane];
#pragma unroll
        for (int i = 0; i < 4; ++i) {
            v16h b = B[((size_t)kt * Ntiles + nt0 + i) * 32 + lane];
            acc[i] = __builtin_amdgcn_wmma_f32_16x16x32_f16(
                false, a_cur, false, b, (short)0, acc[i], false, false);
        }
        a_cur = a_nxt;
    }
#endif

    int lo = lane & 15, hi = lane >> 4;
    int row0 = mt * 16 + hi * 8;
#pragma unroll
    for (int i = 0; i < 4; ++i) {
        int col = (nt0 + i) * 16 + lo;
        float bv = bias ? bias[col] : 0.f;
#pragma unroll
        for (int v = 0; v < 8; ++v) {
            int row = row0 + v;
            float val = acc[i][v] + bv;
            if (res) val += res[(size_t)row * CDIM + col];
            out[(size_t)row * CDIM + col] = val;
        }
    }
}

// ---------------------------------------------------------------------------
// Flash attention: one wave per (b,h, 16-query tile). Online softmax over
// 32-key tiles; P re-layouted C-frag -> A-frag through LDS. K/V fragments of
// the next key tile are prefetched so loads overlap the softmax VALU work.
// ---------------------------------------------------------------------------
__global__ __launch_bounds__(32) void flash_attn(
    const _Float16* __restrict__ Qh, const _Float16* __restrict__ KhT,
    const _Float16* __restrict__ Vh, const float* __restrict__ pbias,
    float* __restrict__ ctx)
{
    __shared__ __align__(16) _Float16 lds[16 * 32];
    int lane = threadIdx.x;
    int lo = lane & 15, hi = lane >> 4;
    int qt = blockIdx.x;
    int bh = blockIdx.y, b = bh >> 3, h = bh & 7;

    const v16h* Q = (const v16h*)(Qh  + (size_t)bh * 256 * 512);
    const v16h* K = (const v16h*)(KhT + (size_t)bh * 256 * 512);
    const v16h* V = (const v16h*)(Vh  + (size_t)bh * 256 * 512);

    v16h aq0 = Q[(size_t)(qt * 2 + 0) * 32 + lane];
    v16h aq1 = Q[(size_t)(qt * 2 + 1) * 32 + lane];

    v8f acc[4] = {};
    float mrow[8], lrow[8];
#pragma unroll
    for (int v = 0; v < 8; ++v) { mrow[v] = -1e30f; lrow[v] = 0.f; }

    float pb[3][3];
#pragma unroll
    for (int i = 0; i < 3; ++i)
#pragma unroll
        for (int j = 0; j < 3; ++j) pb[i][j] = pbias[i * 3 + j];

    for (int kt32 = 0; kt32 < SEQ / 32; ++kt32) {
        // ---- scores: two 16x16 tiles over dk=64 (two K-steps each) --------
        v8f s0 = {}, s1 = {};
        s0 = __builtin_amdgcn_wmma_f32_16x16x32_f16(false, aq0, false,
                K[(size_t)(0 * 128 + kt32 * 2 + 0) * 32 + lane], (short)0, s0, false, false);
        s0 = __builtin_amdgcn_wmma_f32_16x16x32_f16(false, aq1, false,
                K[(size_t)(1 * 128 + kt32 * 2 + 0) * 32 + lane], (short)0, s0, false, false);
        s1 = __builtin_amdgcn_wmma_f32_16x16x32_f16(false, aq0, false,
                K[(size_t)(0 * 128 + kt32 * 2 + 1) * 32 + lane], (short)0, s1, false, false);
        s1 = __builtin_amdgcn_wmma_f32_16x16x32_f16(false, aq1, false,
                K[(size_t)(1 * 128 + kt32 * 2 + 1) * 32 + lane], (short)0, s1, false, false);

        // prefetch next key tile's K/V streams (overlaps softmax below)
        if (kt32 + 1 < SEQ / 32) {
            __builtin_prefetch(&K[(size_t)(0 * 128 + (kt32 + 1) * 2) * 32 + lane], 0, 3);
            __builtin_prefetch(&K[(size_t)(1 * 128 + (kt32 + 1) * 2) * 32 + lane], 0, 3);
            __builtin_prefetch(&V[(size_t)((kt32 + 1) * 4) * 32 + lane], 0, 3);
            __builtin_prefetch(&V[(size_t)((kt32 + 1) * 4 + 2) * 32 + lane], 0, 3);
        }

        int c0 = (kt32 * 32 + lo) % 3;      // phase of key col (this lane, j=0)
        int c1 = (c0 + 1) % 3;              // +16 keys -> phase+1 (16%3==1)
#pragma unroll
        for (int v = 0; v < 8; ++v) {
            int row = qt * 16 + hi * 8 + v;
            int rp = row % 3;
            float sv0 = s0[v] * SCALE + pb[rp][c0];
            float sv1 = s1[v] * SCALE + pb[rp][c1];
            float mx = fmaxf(sv0, sv1);
#pragma unroll
            for (int off = 1; off < 16; off <<= 1)
                mx = fmaxf(mx, __shfl_xor(mx, off, 16));
            float mnew = fmaxf(mrow[v], mx);
            float f  = __expf(mrow[v] - mnew);
            float p0 = __expf(sv0 - mnew);
            float p1 = __expf(sv1 - mnew);
            float ps = p0 + p1;
#pragma unroll
            for (int off = 1; off < 16; off <<= 1)
                ps += __shfl_xor(ps, off, 16);
            lrow[v] = lrow[v] * f + ps;
            mrow[v] = mnew;
#pragma unroll
            for (int nt = 0; nt < 4; ++nt) acc[nt][v] *= f;
            // stage P (C-layout element -> LDS row-major 16x32)
            lds[(hi * 8 + v) * 32 + lo]      = (_Float16)p0;
            lds[(hi * 8 + v) * 32 + 16 + lo] = (_Float16)p1;
        }
        asm volatile("s_wait_dscnt 0x0" ::: "memory");
        // ---- reload P as an A-fragment (lane=row, contiguous K chunks) ----
        v8h pa = *(const v8h*)&lds[lo * 32 + hi * 16];
        v8h pc = *(const v8h*)&lds[lo * 32 + hi * 16 + 8];
        v16h aP;
#pragma unroll
        for (int j = 0; j < 8; ++j) { aP[j] = pa[j]; aP[8 + j] = pc[j]; }
        // ---- ctx += P x V ------------------------------------------------
#pragma unroll
        for (int nt = 0; nt < 4; ++nt) {
            acc[nt] = __builtin_amdgcn_wmma_f32_16x16x32_f16(
                false, aP, false, V[(size_t)(kt32 * 4 + nt) * 32 + lane],
                (short)0, acc[nt], false, false);
        }
    }
    // ---- normalize and scatter to ctx[b, t, h*dk+d] -----------------------
#pragma unroll
    for (int nt = 0; nt < 4; ++nt) {
        int d = nt * 16 + lo;
#pragma unroll
        for (int v = 0; v < 8; ++v) {
            int t = qt * 16 + hi * 8 + v;
            ctx[((size_t)(b * SEQ + t)) * CDIM + h * DK + d] = acc[nt][v] / lrow[v];
        }
    }
}

// ---------------------------------------------------------------------------
// LayerNorm over last dim (512) : one block per row
// ---------------------------------------------------------------------------
__global__ __launch_bounds__(256) void layernorm_k(
    const float* __restrict__ y, const float* __restrict__ g,
    const float* __restrict__ bt, float* __restrict__ out)
{
    __shared__ float rs[8], rq[8];
    int row = blockIdx.x, tid = threadIdx.x;
    float v0 = y[(size_t)row * CDIM + tid];
    float v1 = y[(size_t)row * CDIM + 256 + tid];
    float s = v0 + v1, q = v0 * v0 + v1 * v1;
#pragma unroll
    for (int off = 1; off < 32; off <<= 1) {
        s += __shfl_xor(s, off, 32);
        q += __shfl_xor(q, off, 32);
    }
    if ((tid & 31) == 0) { rs[tid >> 5] = s; rq[tid >> 5] = q; }
    __syncthreads();
    s = 0.f; q = 0.f;
#pragma unroll
    for (int w = 0; w < 8; ++w) { s += rs[w]; q += rq[w]; }
    float mu = s * (1.0f / CDIM);
    float var = q * (1.0f / CDIM) - mu * mu;
    float inv = rsqrtf(var + LN_EPS);
    out[(size_t)row * CDIM + tid]       = (v0 - mu) * inv * g[tid] + bt[tid];
    out[(size_t)row * CDIM + 256 + tid] = (v1 - mu) * inv * g[tid + 256] + bt[tid + 256];
}

// ---------------------------------------------------------------------------
extern "C" void kernel_launch(void* const* d_in, const int* in_sizes, int n_in,
                              void* d_out, int out_size, void* d_ws, size_t ws_size,
                              hipStream_t stream)
{
    const float* x    = (const float*)d_in[0];
    const float* Wq   = (const float*)d_in[1];
    const float* Wq_b = (const float*)d_in[2];
    const float* Wk   = (const float*)d_in[3];
    const float* Wk_b = (const float*)d_in[4];
    const float* Wv   = (const float*)d_in[5];
    const float* Wv_b = (const float*)d_in[6];
    const float* Wo   = (const float*)d_in[7];
    const float* Wo_b = (const float*)d_in[8];
    const float* pbias= (const float*)d_in[9];
    const float* ln_g = (const float*)d_in[10];
    const float* ln_b = (const float*)d_in[11];
    float* outp = (float*)d_out;

    char* p = (char*)d_ws;
    auto alloc = [&](size_t bytes) {
        void* r = (void*)p;
        p += (bytes + 255) & ~(size_t)255;
        return r;
    };
    const size_t TILE_B = 1024;                       // bytes per f16 tile
    _Float16* xh   = (_Float16*)alloc((size_t)8192 * TILE_B); // x A-tiles
    _Float16* wqp  = (_Float16*)alloc((size_t)512 * TILE_B);
    _Float16* wkp  = (_Float16*)alloc((size_t)512 * TILE_B);
    _Float16* wvp  = (_Float16*)alloc((size_t)512 * TILE_B);
    _Float16* wop  = (_Float16*)alloc((size_t)512 * TILE_B);
    float* qf      = (float*)alloc((size_t)MROWS * CDIM * 4);
    float* kf      = (float*)alloc((size_t)MROWS * CDIM * 4);
    float* vf      = (float*)alloc((size_t)MROWS * CDIM * 4);
    _Float16* qh   = (_Float16*)alloc((size_t)32 * 256 * TILE_B);
    _Float16* khT  = (_Float16*)alloc((size_t)32 * 256 * TILE_B);
    _Float16* vh   = (_Float16*)alloc((size_t)32 * 256 * TILE_B);
    float* ctx     = (float*)alloc((size_t)MROWS * CDIM * 4);
    _Float16* ctxh = (_Float16*)alloc((size_t)8192 * TILE_B);
    float* yf      = (float*)alloc((size_t)MROWS * CDIM * 4);

    // 1) pack x and weights to f16 fragment tiles
    pack_a_full<<<2048, 128, 0, stream>>>(x, xh, /*Ktiles=*/16, /*ld=*/CDIM);
    pack_b_weight<<<128, 128, 0, stream>>>(Wq, wqp);
    pack_b_weight<<<128, 128, 0, stream>>>(Wk, wkp);
    pack_b_weight<<<128, 128, 0, stream>>>(Wv, wvp);
    pack_b_weight<<<128, 128, 0, stream>>>(Wo, wop);

    // 2) QKV projections (WMMA + TDM-staged B panels)
    dim3 ggrid(CDIM / 64, MROWS / 64);                // (8, 128)
    gemm_wmma<<<ggrid, 128, 0, stream>>>(xh, wqp, Wq_b, nullptr, qf);
    gemm_wmma<<<ggrid, 128, 0, stream>>>(xh, wkp, Wk_b, nullptr, kf);
    gemm_wmma<<<ggrid, 128, 0, stream>>>(xh, wvp, Wv_b, nullptr, vf);

    // 3) per-head fragment packs
    dim3 hgrid(64, 32);
    pack_q_heads <<<hgrid, 128, 0, stream>>>(qf, qh);
    pack_kT_heads<<<hgrid, 128, 0, stream>>>(kf, khT);
    pack_v_heads <<<hgrid, 128, 0, stream>>>(vf, vh);

    // 4) flash attention (WMMA scores + WMMA PV, online softmax)
    flash_attn<<<dim3(SEQ / 16, 32), 32, 0, stream>>>(qh, khT, vh, pbias, ctx);

    // 5) output projection + bias + residual (WMMA)
    pack_a_full<<<2048, 128, 0, stream>>>(ctx, ctxh, 16, CDIM);
    gemm_wmma<<<ggrid, 128, 0, stream>>>(ctxh, wop, Wo_b, x, yf);

    // 6) LayerNorm
    layernorm_k<<<MROWS, 256, 0, stream>>>(yf, ln_g, ln_b, outp);
}
